// MultiHeadAttention_7112465842555
// MI455X (gfx1250) — matrix operand order repaired and checked by reading, not compiled, not run on hardware
//
#include <hip/hip_runtime.h>

// MI455X / gfx1250 flash-attention MHA forward, bf16 WMMA with fp32 accumulate.
// B=4, T=2048, C=1024, H=16, hs=64. Compute-bound (~138 GFLOP vs ~100MB traffic).
// Attention K/V chunks are staged into LDS with double-buffered ASYNC copies
// (global_load_async_to_lds_b128, ASYNCcnt-tracked) shared by all 4 waves.

#define BDIM  4
#define TSEQ  2048
#define CEMB  1024
#define NHEAD 16
#define HSZ   64

typedef __bf16 bf16_t;
typedef __attribute__((ext_vector_type(16))) __bf16 v16bf;
typedef __attribute__((ext_vector_type(8)))  __bf16 v8bf;
typedef __attribute__((ext_vector_type(8)))  float  v8f;
typedef __attribute__((ext_vector_type(4)))  float  v4f;

// 16-byte async global->LDS copy (tracked by ASYNCcnt). VDST operand is the
// LDS byte address: for generic pointers to LDS, addr[31:0] IS the LDS address
// (aperture bits live in addr[63:32]), so a 32-bit truncation is exact.
static __device__ inline void copy16_g2l(const bf16_t* gsrc, bf16_t* ldst) {
  unsigned lds_addr = (unsigned)(unsigned long long)ldst;
  asm volatile("global_load_async_to_lds_b128 %0, %1, off"
               :
               : "v"(lds_addr), "v"(gsrc)
               : "memory");
}

#define WAIT_ASYNC_8() asm volatile("s_wait_asynccnt 0x8" ::: "memory")
#define WAIT_ASYNC_0() asm volatile("s_wait_asynccnt 0x0" ::: "memory")

static __device__ inline v8f zero8() {
  v8f z;
#pragma unroll
  for (int e = 0; e < 8; e++) z[e] = 0.0f;
  return z;
}

static __device__ inline v8f wmma_bf16(v16bf a, v16bf b, v8f c) {
  // (neg_a, A, neg_b, B, c_mod, C, reuse_a, reuse_b)
  return __builtin_amdgcn_wmma_f32_16x16x32_bf16(false, a, false, b, (short)0, c,
                                                 false, false);
}

// A-fragment (16x32 MxK) from bf16 row-major storage with row stride ld.
// Lane l: row = row0 + (l&15); K chunks at klo={0|8}: [klo..klo+7], [klo+16..klo+23].
static __device__ inline v16bf load_a_bf16(const bf16_t* base, int ld, int row0,
                                           int k0, int lane) {
  const bf16_t* p = base + (size_t)(row0 + (lane & 15)) * ld + k0 + ((lane >> 4) << 3);
  union { v16bf v; v8bf h[2]; } u;
  u.h[0] = *(const v8bf*)p;         // 16B aligned
  u.h[1] = *(const v8bf*)(p + 16);  // 16B aligned
  return u.v;
}

// Same A-fragment but sourced from fp32 row-major, converting to bf16.
static __device__ inline v16bf load_a_f32(const float* base, int ld, int row0,
                                          int k0, int lane) {
  const float* p = base + (size_t)(row0 + (lane & 15)) * ld + k0 + ((lane >> 4) << 3);
  v4f a0 = *(const v4f*)p;
  v4f a1 = *(const v4f*)(p + 4);
  v4f b0 = *(const v4f*)(p + 16);
  v4f b1 = *(const v4f*)(p + 20);
  v16bf r;
#pragma unroll
  for (int i = 0; i < 4; i++) {
    r[i]      = (bf16_t)a0[i];
    r[i + 4]  = (bf16_t)a1[i];
    r[i + 8]  = (bf16_t)b0[i];
    r[i + 12] = (bf16_t)b1[i];
  }
  return r;
}

// B-fragment (32x16 KxN) from N-major ("transposed") storage Bt[n][k], row stride ldk.
// Lane l: N = n0 + (l&15); K chunks at klo={0|8}: [klo..klo+7], [klo+16..klo+23].
static __device__ inline v16bf load_bt_bf16(const bf16_t* bt, int ldk, int n0,
                                            int k0, int lane) {
  const bf16_t* p = bt + (size_t)(n0 + (lane & 15)) * ldk + k0 + ((lane >> 4) << 3);
  union { v16bf v; v8bf h[2]; } u;
  u.h[0] = *(const v8bf*)p;
  u.h[1] = *(const v8bf*)(p + 16);
  return u.v;
}

// ---------------------------------------------------------------------------
// Kernel 0: transpose + convert weights to bf16 N-major layouts.
//   WqT/WkT/WvT[h][d][c] = W[h][c][d]     (B-frag for X@W needs d-major)
//   WpT[n][c]            = Wp[c][n]
// ---------------------------------------------------------------------------
__global__ void prep_weights(const float* __restrict__ Wq, const float* __restrict__ Wk,
                             const float* __restrict__ Wv, const float* __restrict__ Wp,
                             bf16_t* __restrict__ WqT, bf16_t* __restrict__ WkT,
                             bf16_t* __restrict__ WvT, bf16_t* __restrict__ WpT) {
  size_t i = (size_t)blockIdx.x * blockDim.x + threadIdx.x;
  const size_t perW = (size_t)NHEAD * CEMB * HSZ;  // 1,048,576
  size_t which = i / perW;
  size_t j = i % perW;
  if (which < 3) {
    const float* src = (which == 0) ? Wq : ((which == 1) ? Wk : Wv);
    bf16_t* dst = (which == 0) ? WqT : ((which == 1) ? WkT : WvT);
    int h = (int)(j / (CEMB * HSZ));
    int r = (int)(j % (CEMB * HSZ));
    int c = r / HSZ;
    int d = r % HSZ;
    dst[((size_t)h * HSZ + d) * CEMB + c] = (bf16_t)src[((size_t)h * CEMB + c) * HSZ + d];
  } else {
    int c = (int)(j / CEMB);
    int n = (int)(j % CEMB);
    WpT[(size_t)n * CEMB + c] = (bf16_t)Wp[(size_t)c * CEMB + n];
  }
}

// ---------------------------------------------------------------------------
// Kernel 1: fused QKV projection. Block = 64 rows of one (b,h); wave = 16 rows.
// One A-fragment of x feeds q,k,v (12 WMMAs per 32-wide k-step per wave).
// Stores q,k row-major [t][d]; v transposed [d][t] (so attention's PV B-frags
// are contiguous loads).
// ---------------------------------------------------------------------------
__global__ __launch_bounds__(128) void qkv_proj(
    const float* __restrict__ x,
    const bf16_t* __restrict__ WqT, const bf16_t* __restrict__ WkT,
    const bf16_t* __restrict__ WvT,
    bf16_t* __restrict__ qb, bf16_t* __restrict__ kb, bf16_t* __restrict__ vTb) {
  const int lane = threadIdx.x & 31;
  const int w = threadIdx.x >> 5;
  const int t0 = blockIdx.x * 64 + w * 16;
  const int h = blockIdx.y;
  const int b = blockIdx.z;

  const float* xb = x + (size_t)b * TSEQ * CEMB;
  const bf16_t* wq = WqT + (size_t)h * HSZ * CEMB;
  const bf16_t* wk = WkT + (size_t)h * HSZ * CEMB;
  const bf16_t* wv = WvT + (size_t)h * HSZ * CEMB;

  v8f aq[4], ak[4], av[4];
#pragma unroll
  for (int n = 0; n < 4; n++) { aq[n] = zero8(); ak[n] = zero8(); av[n] = zero8(); }

#pragma unroll 1
  for (int c0 = 0; c0 < CEMB; c0 += 32) {
    v16bf a = load_a_f32(xb, CEMB, t0, c0, lane);
#pragma unroll
    for (int n = 0; n < 4; n++) {
      aq[n] = wmma_bf16(a, load_bt_bf16(wq, CEMB, n * 16, c0, lane), aq[n]);
      ak[n] = wmma_bf16(a, load_bt_bf16(wk, CEMB, n * 16, c0, lane), ak[n]);
      av[n] = wmma_bf16(a, load_bt_bf16(wv, CEMB, n * 16, c0, lane), av[n]);
    }
  }

  // C/D layout: element e of acc -> row = mrow+e, col = n*16 + ncol
  const int mrow = (lane >> 4) << 3;
  const int ncol = lane & 15;
  const size_t bh = (size_t)b * NHEAD + h;
  bf16_t* qo = qb + bh * TSEQ * HSZ;
  bf16_t* ko = kb + bh * TSEQ * HSZ;
  bf16_t* vo = vTb + bh * HSZ * TSEQ;
#pragma unroll
  for (int n = 0; n < 4; n++) {
#pragma unroll
    for (int e = 0; e < 8; e++) {
      int t = t0 + mrow + e;
      int d = n * 16 + ncol;
      qo[(size_t)t * HSZ + d] = (bf16_t)aq[n][e];
      ko[(size_t)t * HSZ + d] = (bf16_t)ak[n][e];
      vo[(size_t)d * TSEQ + t] = (bf16_t)av[n][e];  // transposed store
    }
  }
}

// ---------------------------------------------------------------------------
// Kernel 2: causal flash attention with double-buffered async K/V staging.
// Block = 64 query rows of one (b,h); wave = 16 rows. Per 64-wide KV chunk:
// 8 WMMAs (QK^T) + 8 WMMAs (PV) per wave, online softmax in registers,
// K chunk [64s x 64d] and V^T chunk [64d x 64s] staged in LDS once per block.
// ---------------------------------------------------------------------------
__global__ __launch_bounds__(128) void flash_attn(
    const bf16_t* __restrict__ qb, const bf16_t* __restrict__ kb,
    const bf16_t* __restrict__ vTb, bf16_t* __restrict__ ob) {
  __shared__ __align__(16) bf16_t kshm[2][64][64];   // [buf][s_local][d]
  __shared__ __align__(16) bf16_t vshm[2][64][64];   // [buf][d][s_local]
  __shared__ __align__(16) bf16_t pshm[4][16][80];   // per-wave P tile

  const int tid = threadIdx.x;
  const int lane = tid & 31;
  const int w = tid >> 5;
  const int h = blockIdx.y;
  const int b = blockIdx.z;
  const int trow0 = blockIdx.x * 64 + w * 16;
  const size_t bh = (size_t)b * NHEAD + h;
  const bf16_t* q = qb + bh * TSEQ * HSZ;
  const bf16_t* k = kb + bh * TSEQ * HSZ;
  const bf16_t* vT = vTb + bh * HSZ * TSEQ;
  const int mrow = (lane >> 4) << 3;
  const int ncol = lane & 15;

  // cooperative stage of one 64-wide chunk: 8x 16B per thread (4 K + 4 V^T)
  auto stage = [&](int s0, int buf) {
#pragma unroll
    for (int j = 0; j < 4; j++) {
      int idx = tid + j * 128;          // 0..511 16B-chunks
      int row = idx >> 3;               // s_local
      int col = (idx & 7) << 3;         // d
      copy16_g2l(k + (size_t)(s0 + row) * HSZ + col, &kshm[buf][row][col]);
    }
#pragma unroll
    for (int j = 0; j < 4; j++) {
      int idx = tid + j * 128;
      int row = idx >> 3;               // d
      int col = (idx & 7) << 3;         // s_local
      copy16_g2l(vT + (size_t)row * TSEQ + s0 + col, &vshm[buf][row][col]);
    }
  };

  // Q fragments for this wave's 16 rows, all of d=0..63, kept in registers.
  v16bf qf0 = load_a_bf16(q, HSZ, trow0, 0, lane);
  v16bf qf1 = load_a_bf16(q, HSZ, trow0, 32, lane);

  v8f oacc[4];
#pragma unroll
  for (int n = 0; n < 4; n++) oacc[n] = zero8();
  float m[8], lsum[8];
#pragma unroll
  for (int e = 0; e < 8; e++) { m[e] = -3.0e38f; lsum[e] = 0.0f; }

  const float scale = 0.03125f;            // C^-0.5 = 1024^-0.5 (reference)
  const int nchunks = blockIdx.x + 1;      // causal: chunks 0 .. t0/64

  stage(0, 0);

  for (int c = 0; c < nchunks; c++) {
    const int cur = c & 1;
    const int s0 = c * 64;
    if (c + 1 < nchunks) {
      stage((c + 1) * 64, cur ^ 1);        // prefetch next chunk into other buffer
      WAIT_ASYNC_8();                      // current chunk's 8 async loads done
    } else {
      WAIT_ASYNC_0();
    }
    __syncthreads();

    // S = Q K^T for a 16x64 tile (4 N-tiles x 2 k-steps over d), K from LDS
    v8f st[4];
#pragma unroll
    for (int n = 0; n < 4; n++) {
      st[n] = zero8();
      st[n] = wmma_bf16(qf0, load_bt_bf16(&kshm[cur][0][0], 64, n * 16, 0, lane), st[n]);
      st[n] = wmma_bf16(qf1, load_bt_bf16(&kshm[cur][0][0], 64, n * 16, 32, lane), st[n]);
    }

    // scale + causal mask (element (row t, col s): mask s > t)
#pragma unroll
    for (int n = 0; n < 4; n++) {
#pragma unroll
      for (int e = 0; e < 8; e++) {
        float sv = st[n][e] * scale;
        int tg = trow0 + mrow + e;
        int sg = s0 + n * 16 + ncol;
        st[n][e] = (sg > tg) ? -3.0e38f : sv;
      }
    }

    // online softmax; a C/D row lives in one 16-lane half at one element e.
    // Fully-masked tiles degrade to exact no-ops (alpha=1, sum=0, P=0).
    float mnew[8], alpha[8];
#pragma unroll
    for (int e = 0; e < 8; e++) {
      float rm = fmaxf(fmaxf(st[0][e], st[1][e]), fmaxf(st[2][e], st[3][e]));
      rm = fmaxf(rm, __shfl_xor(rm, 8, 32));
      rm = fmaxf(rm, __shfl_xor(rm, 4, 32));
      rm = fmaxf(rm, __shfl_xor(rm, 2, 32));
      rm = fmaxf(rm, __shfl_xor(rm, 1, 32));
      mnew[e] = fmaxf(m[e], rm);
      alpha[e] = __expf(m[e] - mnew[e]);
      m[e] = mnew[e];
    }
#pragma unroll
    for (int n = 0; n < 4; n++)
#pragma unroll
      for (int e = 0; e < 8; e++) st[n][e] = __expf(st[n][e] - mnew[e]);
#pragma unroll
    for (int e = 0; e < 8; e++) {
      float rs = st[0][e] + st[1][e] + st[2][e] + st[3][e];
      rs += __shfl_xor(rs, 8, 32);
      rs += __shfl_xor(rs, 4, 32);
      rs += __shfl_xor(rs, 2, 32);
      rs += __shfl_xor(rs, 1, 32);
      lsum[e] = lsum[e] * alpha[e] + rs;
#pragma unroll
      for (int n = 0; n < 4; n++) oacc[n][e] *= alpha[e];
    }

    // remap P: C/D layout -> row-major bf16 in this wave's private LDS tile
#pragma unroll
    for (int n = 0; n < 4; n++)
#pragma unroll
      for (int e = 0; e < 8; e++)
        pshm[w][mrow + e][n * 16 + ncol] = (bf16_t)st[n][e];
    asm volatile("s_wait_dscnt 0" ::: "memory");  // wave-private tile

    // O += P * V  (A-frags from pshm, B-frags from staged V^T chunk)
#pragma unroll
    for (int ks = 0; ks < 2; ks++) {
      v16bf pf = load_a_bf16(&pshm[w][0][0], 80, 0, ks * 32, lane);
#pragma unroll
      for (int n = 0; n < 4; n++)
        oacc[n] = wmma_bf16(
            pf, load_bt_bf16(&vshm[cur][0][0], 64, n * 16, ks * 32, lane), oacc[n]);
    }
    __syncthreads();  // all waves done with buffer `cur` before it is re-staged
  }

  // normalize and store (concat-heads layout [B,T,C], head offset h*64)
  bf16_t* op = ob + (size_t)b * TSEQ * CEMB + (size_t)h * HSZ;
#pragma unroll
  for (int n = 0; n < 4; n++)
#pragma unroll
    for (int e = 0; e < 8; e++) {
      int t = trow0 + mrow + e;
      int d = n * 16 + ncol;
      op[(size_t)t * CEMB + d] = (bf16_t)(oacc[n][e] / lsum[e]);
    }
}

// ---------------------------------------------------------------------------
// Kernel 3: output projection Out = O @ Wp + bp, fp32 result.
// ---------------------------------------------------------------------------
__global__ __launch_bounds__(128) void out_proj(
    const bf16_t* __restrict__ ob, const bf16_t* __restrict__ WpT,
    const float* __restrict__ bp, float* __restrict__ out) {
  const int lane = threadIdx.x & 31;
  const int w = threadIdx.x >> 5;
  const int m0 = blockIdx.x * 64 + w * 16;
  const int n0 = blockIdx.y * 64;

  v8f acc[4];
#pragma unroll
  for (int n = 0; n < 4; n++) acc[n] = zero8();

#pragma unroll 1
  for (int c0 = 0; c0 < CEMB; c0 += 32) {
    v16bf a = load_a_bf16(ob, CEMB, m0, c0, lane);
#pragma unroll
    for (int n = 0; n < 4; n++)
      acc[n] = wmma_bf16(a, load_bt_bf16(WpT, CEMB, n0 + n * 16, c0, lane), acc[n]);
  }

  const int mrow = (lane >> 4) << 3;
  const int ncol = lane & 15;
#pragma unroll
  for (int n = 0; n < 4; n++)
#pragma unroll
    for (int e = 0; e < 8; e++) {
      int r = m0 + mrow + e;
      int cc = n0 + n * 16 + ncol;
      out[(size_t)r * CEMB + cc] = acc[n][e] + bp[cc];
    }
}

// ---------------------------------------------------------------------------
extern "C" void kernel_launch(void* const* d_in, const int* in_sizes, int n_in,
                              void* d_out, int out_size, void* d_ws, size_t ws_size,
                              hipStream_t stream) {
  const float* x  = (const float*)d_in[0];
  const float* Wq = (const float*)d_in[1];
  const float* Wk = (const float*)d_in[2];
  const float* Wv = (const float*)d_in[3];
  const float* Wp = (const float*)d_in[4];
  const float* bp = (const float*)d_in[5];
  float* out = (float*)d_out;

  // Workspace layout (bf16 elements). Total = 4*8388608 + 4*1048576 elems
  // = 37,748,736 * 2B ≈ 75.5 MB of d_ws.
  bf16_t* ws = (bf16_t*)d_ws;
  const size_t QKV = (size_t)BDIM * NHEAD * TSEQ * HSZ;  // 8,388,608
  const size_t WSZ = (size_t)NHEAD * CEMB * HSZ;         // 1,048,576
  bf16_t* qb   = ws;
  bf16_t* kbuf = qb + QKV;
  bf16_t* vTb  = kbuf + QKV;
  bf16_t* obuf = vTb + QKV;
  bf16_t* WqT  = obuf + QKV;
  bf16_t* WkT  = WqT + WSZ;
  bf16_t* WvT  = WkT + WSZ;
  bf16_t* WpT  = WvT + WSZ;

  const size_t prepElems = 3 * WSZ + (size_t)CEMB * CEMB;  // 4,194,304
  prep_weights<<<(unsigned)(prepElems / 256), 256, 0, stream>>>(
      Wq, Wk, Wv, Wp, WqT, WkT, WvT, WpT);
  qkv_proj<<<dim3(TSEQ / 64, NHEAD, BDIM), 128, 0, stream>>>(
      x, WqT, WkT, WvT, qb, kbuf, vTb);
  flash_attn<<<dim3(TSEQ / 64, NHEAD, BDIM), 128, 0, stream>>>(
      qb, kbuf, vTb, obuf);
  out_proj<<<dim3((BDIM * TSEQ) / 64, CEMB / 64), 128, 0, stream>>>(
      obuf, WpT, bp, out);
}
